// Decoder_33861522162462
// MI455X (gfx1250) — matrix operand order repaired and checked by reading, not compiled, not run on hardware
//
#include <hip/hip_runtime.h>
#include <hip/hip_bf16.h>

typedef __attribute__((ext_vector_type(16))) __bf16 v16bf;
typedef __attribute__((ext_vector_type(8)))  float  v8f;

#define B_   32
#define T_   128
#define V_   32000
#define D_   256
#define U_   512
#define BT   (B_*T_)      // 4096
#define G4   2048         // 4*U
#define NPAD 32064        // (V+1) padded to multiple of 64

// ---------------- conversion / gather / pack kernels ----------------

__global__ void cvt_f32_bf16(const float* __restrict__ src, __bf16* __restrict__ dst, int n) {
    int i = blockIdx.x * blockDim.x + threadIdx.x;
    if (i < n) dst[i] = (__bf16)src[i];
}

// Pack B[K][N] f32 -> Bp[K/2][N] u32 with (bf16(B[2p][n]) , bf16(B[2p+1][n])) pairs.
// Columns >= Nsrc are zero-padded. This matches the WMMA 16-bit B fragment layout:
// each lane needs (K=kp, K=kp+1) for its column, which becomes one 32-bit load.
__global__ void pack_b_pairs(const float* __restrict__ src, uint32_t* __restrict__ dst,
                             int Khalf, int N, int Nsrc) {
    long long i = (long long)blockIdx.x * blockDim.x + threadIdx.x;
    long long total = (long long)Khalf * N;
    if (i >= total) return;
    int p = (int)(i / N);
    int n = (int)(i % N);
    float lo = (n < Nsrc) ? src[(long long)(2 * p)     * Nsrc + n] : 0.f;
    float hi = (n < Nsrc) ? src[(long long)(2 * p + 1) * Nsrc + n] : 0.f;
    union { uint32_t u; __bf16 h[2]; } cv;
    cv.h[0] = (__bf16)lo;
    cv.h[1] = (__bf16)hi;
    dst[i] = cv.u;
}

__global__ void embed_gather_bf16(const int* __restrict__ tok, const float* __restrict__ E,
                                  __bf16* __restrict__ x) {
    int i = blockIdx.x * blockDim.x + threadIdx.x;
    if (i >= BT * D_) return;
    int r = i / D_, d = i % D_;
    x[i] = (__bf16)E[(long long)tok[r] * D_ + d];
}

// ---------------- bf16 WMMA GEMM: C[M,N] = A[M,K] @ B[K,N] (+bias) ----------------
// A row-major bf16 [M][K]; Bp pair-packed u32 [K/2][N]; C f32, leading dim ldc,
// stores only columns < Nstore. One wave computes a 16x64 strip (4 C tiles),
// reusing the A fragment across 4 WMMAs per k-step. 8 waves / block.

template <bool HAS_BIAS>
__global__ __launch_bounds__(256) void gemm_bf16_wmma(
    const __bf16* __restrict__ A, const uint32_t* __restrict__ Bp,
    float* __restrict__ C, const float* __restrict__ bias,
    int M, int N, int K, int ldc, int Nstore)
{
    int wave = threadIdx.x >> 5;
    int lane = threadIdx.x & 31;
    int tilesN = N >> 6;                    // 64-wide strips
    long long strip = (long long)blockIdx.x * 8 + wave;
    if (strip >= (long long)(M >> 4) * tilesN) return;
    int tm = (int)(strip / tilesN);
    int tn = (int)(strip % tilesN);
    int half = lane >> 4;
    int l16  = lane & 15;

    v8f acc[4];
#pragma unroll
    for (int j = 0; j < 4; ++j) acc[j] = (v8f){0.f,0.f,0.f,0.f,0.f,0.f,0.f,0.f};

    const __bf16*   Arow = A  + (long long)(tm * 16 + l16) * K;
    int colbase = tn * 64 + l16;

    for (int kk = 0; kk < K; kk += 32) {
        // A fragment (contiguous k order k = 16*half + s, same as B): all K pairs are
        // even-based and contiguous -> 8 dword loads.
        v16bf af;
#pragma unroll
        for (int i = 0; i < 8; ++i) {
            int kb = 2 * i + half * 16;
            union { uint32_t u; __bf16 h[2]; } cv;
            cv.u = *(const uint32_t*)(Arow + kk + kb);
            af[2 * i]     = cv.h[0];
            af[2 * i + 1] = cv.h[1];
        }
        // 4 B fragments from the pair-packed layout: pair row = kk/2 + i + half*8
        const uint32_t* Brow = Bp + (long long)((kk >> 1) + half * 8) * N + colbase;
#pragma unroll
        for (int j = 0; j < 4; ++j) {
            v16bf bfrag;
#pragma unroll
            for (int i = 0; i < 8; ++i) {
                union { uint32_t u; __bf16 h[2]; } cv;
                cv.u = Brow[(long long)i * N + j * 16];
                bfrag[2 * i]     = cv.h[0];
                bfrag[2 * i + 1] = cv.h[1];
            }
            acc[j] = __builtin_amdgcn_wmma_f32_16x16x32_bf16(
                false, af, false, bfrag, (short)0, acc[j], false, false);
        }
    }

    // Epilogue: M is always a multiple of 16 here -> no row guard, strided stores.
    float* Cbase = C + (size_t)(tm * 16 + half * 8) * ldc;
#pragma unroll
    for (int j = 0; j < 4; ++j) {
        int col = colbase + j * 16;
        if (col < Nstore) {
            float bv = HAS_BIAS ? bias[col] : 0.f;
            float* Cp = Cbase + col;
#pragma unroll
            for (int r = 0; r < 8; ++r)
                Cp[(size_t)r * ldc] = acc[j][r] + bv;
        }
    }
}

// ---------------- persistent per-batch-row recurrence ----------------

__device__ __forceinline__ float sigf(float x) { return 1.f / (1.f + __expf(-x)); }

__global__ __launch_bounds__(512) void decoder_scan(
    const float* __restrict__ xw,    // [BT][2048]  x@W_lstm[:D] + b (precomputed)
    const float* __restrict__ Wl2,   // [512][2048] = W_lstm rows D.. (f32)
    const float* __restrict__ Ul,    // [512][2048] U_lstm (f32)
    const float* __restrict__ keys,  // [BT][512]
    const float* __restrict__ mem,   // [BT][512]
    const float* __restrict__ Wa,    // [1024][512]
    const float* __restrict__ h0,
    const float* __restrict__ c0,
    __bf16* __restrict__ attnA)      // [BT][512] bf16
{
    __shared__ float h[U_], c[U_], at[U_], cx[U_], sc[T_], red[64];
    int b = blockIdx.x;
    int tid = threadIdx.x;

    h[tid]  = h0[b * U_ + tid];
    c[tid]  = c0[b * U_ + tid];
    at[tid] = 0.f;
    __syncthreads();

    for (int t = 0; t < T_; ++t) {
        const float* xwr = xw + (long long)(b * T_ + t) * G4;
        float zi = xwr[tid], zf = xwr[512 + tid], zg = xwr[1024 + tid], zo = xwr[1536 + tid];
#pragma unroll 4
        for (int k = 0; k < U_; ++k) {
            float a = at[k], hh = h[k];
            const float* w = Wl2 + (long long)k * G4;
            const float* u = Ul  + (long long)k * G4;
            zi += a * w[tid]        + hh * u[tid];
            zf += a * w[512 + tid]  + hh * u[512 + tid];
            zg += a * w[1024 + tid] + hh * u[1024 + tid];
            zo += a * w[1536 + tid] + hh * u[1536 + tid];
        }
        float cn = sigf(zf) * c[tid] + sigf(zi) * tanhf(zg);
        float hn = sigf(zo) * tanhf(cn);
        __syncthreads();
        c[tid] = cn; h[tid] = hn;
        __syncthreads();

        if (tid < T_) {
            float s = 0.f;
            const float* kr = keys + (long long)(b * T_ + tid) * U_;
            for (int k = 0; k < U_; ++k) s += h[k] * kr[k];
            sc[tid] = s;
        }
        __syncthreads();

        if (tid < 64) red[tid] = fmaxf(sc[tid], sc[tid + 64]);
        __syncthreads();
        if (tid < 32) red[tid] = fmaxf(red[tid], red[tid + 32]);
        __syncthreads();
        if (tid < 16) red[tid] = fmaxf(red[tid], red[tid + 16]);
        __syncthreads();
        if (tid < 8)  red[tid] = fmaxf(red[tid], red[tid + 8]);
        __syncthreads();
        if (tid < 4)  red[tid] = fmaxf(red[tid], red[tid + 4]);
        __syncthreads();
        if (tid < 2)  red[tid] = fmaxf(red[tid], red[tid + 2]);
        __syncthreads();
        if (tid == 0) red[0] = fmaxf(red[0], red[1]);
        __syncthreads();
        float mx = red[0];
        if (tid < T_) sc[tid] = __expf(sc[tid] - mx);
        __syncthreads();
        if (tid < 64) red[tid] = sc[tid] + sc[tid + 64];
        __syncthreads();
        if (tid < 32) red[tid] += red[tid + 32];
        __syncthreads();
        if (tid < 16) red[tid] += red[tid + 16];
        __syncthreads();
        if (tid < 8)  red[tid] += red[tid + 8];
        __syncthreads();
        if (tid < 4)  red[tid] += red[tid + 4];
        __syncthreads();
        if (tid < 2)  red[tid] += red[tid + 2];
        __syncthreads();
        if (tid == 0) red[0] += red[1];
        __syncthreads();
        float inv = 1.f / red[0];
        if (tid < T_) sc[tid] *= inv;
        __syncthreads();

        {
            float s = 0.f;
            for (int t2 = 0; t2 < T_; ++t2)
                s += sc[t2] * mem[(long long)(b * T_ + t2) * U_ + tid];
            cx[tid] = s;
        }
        __syncthreads();

        float an = 0.f;
        for (int k = 0; k < U_; ++k)
            an += h[k] * Wa[(long long)k * U_ + tid]
                + cx[k] * Wa[(long long)(U_ + k) * U_ + tid];
        __syncthreads();
        at[tid] = an;
        attnA[(long long)(b * T_ + t) * U_ + tid] = (__bf16)an;
        __syncthreads();
    }
}

// ---------------- launch ----------------

extern "C" void kernel_launch(void* const* d_in, const int* in_sizes, int n_in,
                              void* d_out, int out_size, void* d_ws, size_t ws_size,
                              hipStream_t stream)
{
    const int*   inputs = (const int*)  d_in[0];
    const float* memory = (const float*)d_in[1];
    const float* enc_h  = (const float*)d_in[2];
    const float* enc_c  = (const float*)d_in[3];
    const float* E      = (const float*)d_in[4];
    const float* Wm     = (const float*)d_in[5];
    const float* W_lstm = (const float*)d_in[6];
    const float* U_lstm = (const float*)d_in[7];
    const float* b_lstm = (const float*)d_in[8];
    const float* Wa     = (const float*)d_in[9];
    const float* Wfc    = (const float*)d_in[10];
    const float* bfc    = (const float*)d_in[11];
    float* out = (float*)d_out;

    char* ws = (char*)d_ws;
    size_t off = 0;
    auto wsalloc = [&](size_t bytes) -> char* {
        char* p = ws + off;
        off += (bytes + 255) & ~(size_t)255;
        return p;
    };
    float*    xw     = (float*)   wsalloc((size_t)BT * G4 * 4);          // 32 MB
    float*    keys   = (float*)   wsalloc((size_t)BT * U_ * 4);          // 8 MB
    __bf16*   xbf    = (__bf16*)  wsalloc((size_t)BT * D_ * 2);          // 2 MB
    __bf16*   membf  = (__bf16*)  wsalloc((size_t)BT * U_ * 2);          // 4 MB
    __bf16*   attnbf = (__bf16*)  wsalloc((size_t)BT * U_ * 2);          // 4 MB
    uint32_t* wmp    = (uint32_t*)wsalloc((size_t)(U_/2) * U_ * 4);      // 0.5 MB
    uint32_t* wlp    = (uint32_t*)wsalloc((size_t)(D_/2) * G4 * 4);      // 1 MB
    uint32_t* wfcp   = (uint32_t*)wsalloc((size_t)(U_/2) * NPAD * 4);    // 33 MB

    const int TPB = 256;

    // A-side bf16 conversions (row-major element order)
    { int n = BT * U_; cvt_f32_bf16<<<(n + TPB - 1) / TPB, TPB, 0, stream>>>(memory, membf, n); }
    { int n = BT * D_; embed_gather_bf16<<<(n + TPB - 1) / TPB, TPB, 0, stream>>>(inputs, E, xbf); }

    // B-side pair-packed conversions
    { long long n = (long long)(U_/2) * U_;
      pack_b_pairs<<<(unsigned)((n + TPB - 1) / TPB), TPB, 0, stream>>>(Wm, wmp, U_/2, U_, U_); }
    { long long n = (long long)(D_/2) * G4;
      pack_b_pairs<<<(unsigned)((n + TPB - 1) / TPB), TPB, 0, stream>>>(W_lstm, wlp, D_/2, G4, G4); } // rows 0..D-1 contiguous
    { long long n = (long long)(U_/2) * NPAD;
      pack_b_pairs<<<(unsigned)((n + TPB - 1) / TPB), TPB, 0, stream>>>(Wfc, wfcp, U_/2, NPAD, V_ + 1); }

    // keys = memory @ Wm   [4096,512]@[512,512]
    {
        long long waves = (long long)(BT / 16) * (U_ / 64);
        gemm_bf16_wmma<false><<<(unsigned)((waves + 7) / 8), 256, 0, stream>>>(
            membf, wmp, keys, nullptr, BT, U_, U_, U_, U_);
    }
    // xw = E[inputs] @ W_lstm[:D] + b_lstm   [4096,256]@[256,2048]
    {
        long long waves = (long long)(BT / 16) * (G4 / 64);
        gemm_bf16_wmma<true><<<(unsigned)((waves + 7) / 8), 256, 0, stream>>>(
            xbf, wlp, xw, b_lstm, BT, G4, D_, G4, G4);
    }

    // Sequential recurrence: one persistent workgroup per batch row
    decoder_scan<<<B_, 512, 0, stream>>>(
        xw, W_lstm + (long long)D_ * G4, U_lstm, keys, memory, Wa, enc_h, enc_c, attnbf);

    // logits = attn @ Wfc + bfc   [4096,512]@[512,32001] -> d_out [B,T,V+1]
    {
        long long waves = (long long)(BT / 16) * (NPAD / 64);
        gemm_bf16_wmma<true><<<(unsigned)((waves + 7) / 8), 256, 0, stream>>>(
            attnbf, wfcp, out, bfc, BT, NPAD, U_, V_ + 1, V_ + 1);
    }
}
